// TemporalROIFusion_33818572489227
// MI455X (gfx1250) — matrix operand order repaired and checked by reading, not compiled, not run on hardware
//
#include <hip/hip_runtime.h>
#include <hip/hip_bf16.h>

typedef __attribute__((ext_vector_type(16))) _Float16 v16h;
typedef __attribute__((ext_vector_type(8)))  float    v8f;
typedef __attribute__((ext_vector_type(4)))  int      v4i;

#define N_ROI 32768
#define DIM   1024
#define NHEAD 8
#define HD    128
#define TOPK  2048

// ---------------------------------------------------------------------------
// CDNA5 async global->LDS copy (ASYNCcnt path), with sync fallback
// builtin signature (from clang-22 diagnostic): (AS1 v4i*, AS3 v4i*, Ii, Ii)
// ---------------------------------------------------------------------------
typedef __attribute__((address_space(1))) v4i* gptr_v4i;
typedef __attribute__((address_space(3))) v4i* lptr_v4i;

#if __has_builtin(__builtin_amdgcn_global_load_async_to_lds_b128)
#define ASYNC_LDS 1
#endif

__device__ __forceinline__ void async_cp16(const _Float16* g, _Float16* l) {
#ifdef ASYNC_LDS
  __builtin_amdgcn_global_load_async_to_lds_b128((gptr_v4i)g, (lptr_v4i)l, 0, 0);
#else
  ((uint4*)l)[0] = ((const uint4*)g)[0];
#endif
}

__device__ __forceinline__ void wait_async_all() {
#ifdef ASYNC_LDS
#if __has_builtin(__builtin_amdgcn_s_wait_asynccnt)
  __builtin_amdgcn_s_wait_asynccnt(0);
#else
  asm volatile("s_wait_asynccnt 0x0" ::: "memory");
#endif
#endif
}

// ---------------------------------------------------------------------------
// Fragment index helpers (CDNA5 16-bit WMMA layouts, CDNA5 ISA)
// A 16x32 f16: lane m=lane&15; half j -> K = j + ((lane>>4)&1)*16 (this code's k order, not AMD's table)
// B 32x16 f16: lane n=lane&15; half j -> K = j + ((lane>>4)&1)*16 (same order as A, not AMD's table)
// C 16x16 f32: vgpr r, lane l -> (M = r + ((l>>4)&1)*8, N = l&15)
// ---------------------------------------------------------------------------
__device__ __forceinline__ int a_k_of(int j, int lane) {
  return j + (((lane >> 4) & 1) << 4);
}

// ---------------------------------------------------------------------------
// 1) scores + monotonic key transform + flag init (bandwidth pass)
// ---------------------------------------------------------------------------
__global__ void scores_keys_kernel(const float* __restrict__ roi,
                                   const float* __restrict__ wsc,
                                   const float* __restrict__ bsc,
                                   unsigned* __restrict__ keys,
                                   int* __restrict__ flag) {
  int wid  = (blockIdx.x * blockDim.x + threadIdx.x) >> 5;  // wave == row
  int lane = threadIdx.x & 31;
  if (wid >= N_ROI) return;
  const float* r = roi + (size_t)wid * DIM;
  float s = 0.f;
#pragma unroll 8
  for (int i = lane; i < DIM; i += 32) s += r[i] * wsc[i];
  for (int m = 16; m >= 1; m >>= 1) s += __shfl_xor(s, m, 32);
  if (lane == 0) {
    s += bsc[0];
    unsigned b = __float_as_uint(s);
    keys[wid]  = (b & 0x80000000u) ? ~b : (b | 0x80000000u);
    flag[wid]  = -1;
  }
}

// ---------------------------------------------------------------------------
// 2) radix select (4 x 8-bit passes)
// ---------------------------------------------------------------------------
__global__ void init_kernel(unsigned* __restrict__ hist, int* __restrict__ state) {
  for (int i = threadIdx.x; i < 1024; i += blockDim.x) hist[i] = 0;
  if (threadIdx.x == 0) { state[0] = 0; state[1] = TOPK; state[2] = 0; }
}

__global__ void hist_kernel(const unsigned* __restrict__ keys,
                            unsigned* __restrict__ hist,
                            const int* __restrict__ state, int pass) {
  __shared__ unsigned h[256];
  for (int i = threadIdx.x; i < 256; i += blockDim.x) h[i] = 0;
  __syncthreads();
  unsigned prefix = (unsigned)state[0];
  int shift = 24 - 8 * pass;
  int n = blockIdx.x * blockDim.x + threadIdx.x;
  if (n < N_ROI) {
    unsigned k = keys[n];
    if (((unsigned long long)(k ^ prefix) >> (shift + 8)) == 0ull)
      atomicAdd(&h[(k >> shift) & 255u], 1u);
  }
  __syncthreads();
  for (int i = threadIdx.x; i < 256; i += blockDim.x)
    if (h[i]) atomicAdd(&hist[pass * 256 + i], h[i]);
}

__global__ void scan_kernel(const unsigned* __restrict__ hist,
                            int* __restrict__ state, int pass) {
  if (threadIdx.x != 0) return;
  unsigned prefix    = (unsigned)state[0];
  unsigned remaining = (unsigned)state[1];
  unsigned total_gt  = (unsigned)state[2];
  int shift = 24 - 8 * pass;
  unsigned cum = 0;
  int d = 255;
  for (; d > 0; --d) {
    unsigned c = hist[pass * 256 + d];
    if (cum + c >= remaining) break;
    cum += c;
  }
  state[0] = (int)(prefix | ((unsigned)d << shift));
  state[1] = (int)(remaining - cum);
  state[2] = (int)(total_gt + cum);
}

// ---------------------------------------------------------------------------
// 3) deterministic compaction (single block, ballot-ordered by index)
// ---------------------------------------------------------------------------
__global__ __launch_bounds__(1024)
void compact_kernel(const unsigned* __restrict__ keys,
                    const int* __restrict__ state,
                    int* __restrict__ idx, int* __restrict__ flag) {
  __shared__ unsigned woff_gt[32], woff_eq[32];
  __shared__ unsigned run_gt, run_eq;
  unsigned T    = (unsigned)state[0];
  unsigned n_gt = (unsigned)state[2];
  int tid = threadIdx.x, lane = tid & 31, wv = tid >> 5;
  unsigned below = (1u << lane) - 1u;
  if (tid == 0) { run_gt = 0; run_eq = 0; }
  __syncthreads();
  for (int it = 0; it < N_ROI / 1024; ++it) {
    int n = it * 1024 + tid;
    unsigned k = keys[n];
    bool gt = k > T, eq = k == T;
    unsigned mgt = (unsigned)__ballot(gt);
    unsigned meq = (unsigned)__ballot(eq);
    if (lane == 0) { woff_gt[wv] = __popc(mgt); woff_eq[wv] = __popc(meq); }
    __syncthreads();
    if (tid == 0) {
      unsigned ag = run_gt, ae = run_eq;
      for (int i = 0; i < 32; ++i) {
        unsigned cg = woff_gt[i]; woff_gt[i] = ag; ag += cg;
        unsigned ce = woff_eq[i]; woff_eq[i] = ae; ae += ce;
      }
      run_gt = ag; run_eq = ae;
    }
    __syncthreads();
    if (gt) {
      unsigned slot = woff_gt[wv] + __popc(mgt & below);
      idx[slot] = n; flag[n] = (int)slot;
    }
    if (eq) {
      unsigned slot = n_gt + woff_eq[wv] + __popc(meq & below);
      if (slot < TOPK) { idx[slot] = n; flag[n] = (int)slot; }
    }
    __syncthreads();
  }
}

// ---------------------------------------------------------------------------
// 4) gather selected rows into right half of A2 = [agg | dst] (f16)
// ---------------------------------------------------------------------------
__global__ void gather_kernel(const float* __restrict__ roi,
                              const int* __restrict__ idx,
                              _Float16* __restrict__ A2) {
  int t = blockIdx.x * blockDim.x + threadIdx.x;
  if (t >= TOPK * DIM / 4) return;
  int j = t / (DIM / 4), d4 = t % (DIM / 4);
  float4 v = ((const float4*)(roi + (size_t)idx[j] * DIM))[d4];
  _Float16* d = A2 + (size_t)j * (2 * DIM) + DIM + d4 * 4;
  d[0] = (_Float16)v.x; d[1] = (_Float16)v.y;
  d[2] = (_Float16)v.z; d[3] = (_Float16)v.w;
}

__global__ void f32_to_f16_kernel(const float* __restrict__ src,
                                  _Float16* __restrict__ dst, int n4) {
  int t = blockIdx.x * blockDim.x + threadIdx.x;
  if (t >= n4) return;
  float4 v = ((const float4*)src)[t];
  dst[4 * t + 0] = (_Float16)v.x; dst[4 * t + 1] = (_Float16)v.y;
  dst[4 * t + 2] = (_Float16)v.z; dst[4 * t + 3] = (_Float16)v.w;
}

// ---------------------------------------------------------------------------
// 5) NT GEMM: C[m,n] = sum_k A[m,k]*W[n,k] + bias[n]
//    128x128 tile, 8 waves; double-buffered LDS + async global->LDS DMA
// ---------------------------------------------------------------------------
__global__ __launch_bounds__(256)
void gemm_nt_kernel(const _Float16* __restrict__ A, int lda,
                    const _Float16* __restrict__ W,
                    const float* __restrict__ bias,
                    void* __restrict__ Cout, int out_f32, int ldc, int Kd) {
  __shared__ __align__(16) _Float16 As[2][128 * 32];
  __shared__ __align__(16) _Float16 Ws[2][128 * 32];
  int m0 = blockIdx.y * 128, n0 = blockIdx.x * 128;
  int tid = threadIdx.x, lane = tid & 31, wv = tid >> 5;
  int ml = lane & 15, hi = (lane >> 4) & 1;
  int rowA = tid >> 1, seg = (tid & 1) * 16;

  const _Float16* gA = A + (size_t)(m0 + rowA) * lda + seg;
  const _Float16* gW = W + (size_t)(n0 + rowA) * Kd + seg;

  auto stage = [&](int buf, int k0) {
    async_cp16(gA + k0,     &As[buf][rowA * 32 + seg]);
    async_cp16(gA + k0 + 8, &As[buf][rowA * 32 + seg + 8]);
    async_cp16(gW + k0,     &Ws[buf][rowA * 32 + seg]);
    async_cp16(gW + k0 + 8, &Ws[buf][rowA * 32 + seg + 8]);
  };

  v8f acc[8] = {};
  stage(0, 0);
  for (int k0 = 0; k0 < Kd; k0 += 32) {
    wait_async_all();
    __syncthreads();
    int buf = (k0 >> 5) & 1;
    if (k0 + 32 < Kd) stage(buf ^ 1, k0 + 32);       // DMA next tile overlaps WMMA
    if (k0 + 64 < Kd) {                              // global_prefetch_b8 two ahead
      __builtin_prefetch(gA + k0 + 64, 0, 1);
      __builtin_prefetch(gW + k0 + 64, 0, 1);
    }
    v16h a;
#pragma unroll
    for (int j = 0; j < 16; ++j) a[j] = As[buf][(wv * 16 + ml) * 32 + a_k_of(j, lane)];
    v16h bfr[8];
#pragma unroll
    for (int nt = 0; nt < 8; ++nt)
#pragma unroll
      for (int j = 0; j < 16; ++j)
        bfr[nt][j] = Ws[buf][(nt * 16 + ml) * 32 + (hi << 4) + j];
#pragma unroll
    for (int nt = 0; nt < 8; ++nt)
      acc[nt] = __builtin_amdgcn_wmma_f32_16x16x32_f16(false, a, false, bfr[nt],
                                                       (short)0, acc[nt], false, false);
  }
#pragma unroll
  for (int nt = 0; nt < 8; ++nt) {
    int n = n0 + nt * 16 + ml;
    float bv = bias[n];
#pragma unroll
    for (int r = 0; r < 8; ++r) {
      int m = m0 + wv * 16 + r + (hi << 3);
      float v = acc[nt][r] + bv;
      if (out_f32) ((float*)Cout)[(size_t)m * ldc + n] = v;
      else         ((_Float16*)Cout)[(size_t)m * ldc + n] = (_Float16)v;
    }
  }
}

// ---------------------------------------------------------------------------
// 6) per-(row,head) L2 norms (+1e-6) of Q/K, one wave each
// ---------------------------------------------------------------------------
__global__ void norms_kernel(const _Float16* __restrict__ X, float* __restrict__ nrm) {
  int wid  = (blockIdx.x * blockDim.x + threadIdx.x) >> 5;
  int lane = threadIdx.x & 31;
  if (wid >= TOPK * NHEAD) return;
  int q = wid >> 3, h = wid & 7;
  const _Float16* p = X + (size_t)q * DIM + h * HD;
  float s = 0.f;
#pragma unroll
  for (int i = lane; i < HD; i += 32) { float v = (float)p[i]; s += v * v; }
  for (int m = 16; m >= 1; m >>= 1) s += __shfl_xor(s, m, 32);
  if (lane == 0) nrm[h * TOPK + q] = sqrtf(s) + 1e-6f;
}

// ---------------------------------------------------------------------------
// 7) flash attention per head with cosine gate; writes agg into A2 left half
//    block = (128 query rows, head); wave = 16 rows; key tile = 32,
//    double-buffered async K/V staging
// ---------------------------------------------------------------------------
__global__ __launch_bounds__(256)
void flash_kernel(const _Float16* __restrict__ Qh, const _Float16* __restrict__ Kh,
                  const _Float16* __restrict__ Vh, const float* __restrict__ nq,
                  const float* __restrict__ nk, _Float16* __restrict__ A2) {
  __shared__ __align__(16) _Float16 Ks[2][32 * 128];
  __shared__ __align__(16) _Float16 Vs[2][32 * 128];
  __shared__ float    Sbuf[8][16 * 32];
  __shared__ _Float16 Pbuf[8][16 * 32];
  __shared__ float    rowbuf[8][16];
  int h = blockIdx.y, q0 = blockIdx.x * 128;
  int tid = threadIdx.x, lane = tid & 31, wv = tid >> 5;
  int ml = lane & 15, hi = (lane >> 4) & 1;
  const float scale = 0.08838834764831845f;  // 1/sqrt(128)

  v16h aq[4];
  int qrow = q0 + wv * 16 + ml;
#pragma unroll
  for (int c = 0; c < 4; ++c)
#pragma unroll
    for (int j = 0; j < 16; ++j)
      aq[c][j] = Qh[(size_t)qrow * DIM + h * HD + c * 32 + a_k_of(j, lane)];

  int rKV = tid >> 3, segKV = (tid & 7) * 16;
  auto stageKV = [&](int buf, int jt) {
    const _Float16* gk = Kh + (size_t)(jt * 32 + rKV) * DIM + h * HD + segKV;
    const _Float16* gv = Vh + (size_t)(jt * 32 + rKV) * DIM + h * HD + segKV;
    async_cp16(gk,     &Ks[buf][rKV * 128 + segKV]);
    async_cp16(gk + 8, &Ks[buf][rKV * 128 + segKV + 8]);
    async_cp16(gv,     &Vs[buf][rKV * 128 + segKV]);
    async_cp16(gv + 8, &Vs[buf][rKV * 128 + segKV + 8]);
  };

  v8f acc[8] = {};
  float mrow = -1e30f, lrow = 0.f, nqv = 1.f;
  if (lane < 16) nqv = nq[h * TOPK + q0 + wv * 16 + lane];

  stageKV(0, 0);
  for (int jt = 0; jt < TOPK / 32; ++jt) {
    wait_async_all();
    __syncthreads();
    int buf = jt & 1;
    if (jt + 1 < TOPK / 32) stageKV(buf ^ 1, jt + 1);  // DMA overlaps compute
    // S = Q K^T (two 16x16 C-tiles -> Sbuf)
#pragma unroll
    for (int t2 = 0; t2 < 2; ++t2) {
      v16h bk4[4];
#pragma unroll
      for (int c = 0; c < 4; ++c)
#pragma unroll
        for (int j = 0; j < 16; ++j)
          bk4[c][j] = Ks[buf][(t2 * 16 + ml) * 128 + c * 32 + (hi << 4) + j];
      v8f s = {};
#pragma unroll
      for (int c = 0; c < 4; ++c)
        s = __builtin_amdgcn_wmma_f32_16x16x32_f16(false, aq[c], false, bk4[c],
                                                   (short)0, s, false, false);
#pragma unroll
      for (int r = 0; r < 8; ++r)
        Sbuf[wv][(r + (hi << 3)) * 32 + t2 * 16 + ml] = s[r];
    }
    // online softmax with cosine gate (lanes 0..15 each own one row)
    if (lane < 16) {
      const float* srow = &Sbuf[wv][lane * 32];
      float mnew = mrow;
#pragma unroll
      for (int j = 0; j < 32; ++j) {
        float sv  = srow[j];
        float sim = sv / (nqv * nk[h * TOPK + jt * 32 + j]);
        if (sim > 0.9f) mnew = fmaxf(mnew, sv * scale);
      }
      float factor = (mrow > -1e29f) ? __expf(mrow - mnew) : 0.f;
      float lsum = lrow * factor;
#pragma unroll
      for (int j = 0; j < 32; ++j) {
        float sv  = srow[j];
        float sim = sv / (nqv * nk[h * TOPK + jt * 32 + j]);
        float pv  = (sim > 0.9f) ? __expf(sv * scale - mnew) : 0.f;
        lsum += pv;
        Pbuf[wv][lane * 32 + j] = (_Float16)pv;
      }
      mrow = mnew; lrow = lsum;
      rowbuf[wv][lane] = factor;
    }
    // rescale acc, then acc += P x V  (wave-local LDS: DS ops are in-order)
#pragma unroll
    for (int r = 0; r < 8; ++r) {
      float fr = rowbuf[wv][r + (hi << 3)];
#pragma unroll
      for (int nt = 0; nt < 8; ++nt) acc[nt][r] *= fr;
    }
    v16h ap;
#pragma unroll
    for (int j = 0; j < 16; ++j) ap[j] = Pbuf[wv][ml * 32 + a_k_of(j, lane)];
    v16h bv8[8];
#pragma unroll
    for (int nt = 0; nt < 8; ++nt)
#pragma unroll
      for (int j = 0; j < 16; ++j)
        bv8[nt][j] = Vs[buf][((hi << 4) + j) * 128 + nt * 16 + ml];
#pragma unroll
    for (int nt = 0; nt < 8; ++nt)
      acc[nt] = __builtin_amdgcn_wmma_f32_16x16x32_f16(false, ap, false, bv8[nt],
                                                       (short)0, acc[nt], false, false);
  }
  // epilogue: normalize (zero rows with no match), write agg to A2[:, h*128..]
  if (lane < 16) rowbuf[wv][lane] = (lrow > 0.f) ? 1.f / lrow : 0.f;
#pragma unroll
  for (int r = 0; r < 8; ++r) {
    float fr = rowbuf[wv][r + (hi << 3)];
    int m = q0 + wv * 16 + r + (hi << 3);
#pragma unroll
    for (int nt = 0; nt < 8; ++nt)
      A2[(size_t)m * (2 * DIM) + h * HD + nt * 16 + ml] = (_Float16)(acc[nt][r] * fr);
  }
}

// ---------------------------------------------------------------------------
// 8) final: out = roi, with selected rows replaced by 0.5*roi + 0.5*fused
// ---------------------------------------------------------------------------
__global__ void finalize_kernel(const float* __restrict__ roi,
                                const int* __restrict__ flag,
                                const float* __restrict__ Obuf,
                                float* __restrict__ out) {
  int t = blockIdx.x * blockDim.x + threadIdx.x;
  if (t >= N_ROI * DIM / 4) return;
  int row = t / (DIM / 4), d4 = t % (DIM / 4);
  float4 v = ((const float4*)roi)[t];
  int f = flag[row];
  if (f >= 0) {
    float4 o = ((const float4*)(Obuf + (size_t)f * DIM))[d4];
    v.x = 0.5f * (v.x + o.x); v.y = 0.5f * (v.y + o.y);
    v.z = 0.5f * (v.z + o.z); v.w = 0.5f * (v.w + o.w);
  }
  ((float4*)out)[t] = v;
}

// ---------------------------------------------------------------------------
extern "C" void kernel_launch(void* const* d_in, const int* in_sizes, int n_in,
                              void* d_out, int out_size, void* d_ws, size_t ws_size,
                              hipStream_t stream) {
  const float* roi   = (const float*)d_in[0];
  const float* w_sc  = (const float*)d_in[1];
  const float* b_sc  = (const float*)d_in[2];
  const float* wq    = (const float*)d_in[3];
  const float* bq    = (const float*)d_in[4];
  const float* wk    = (const float*)d_in[5];
  const float* bk    = (const float*)d_in[6];
  const float* wv    = (const float*)d_in[7];
  const float* bv    = (const float*)d_in[8];
  const float* wo    = (const float*)d_in[9];
  const float* bo    = (const float*)d_in[10];
  float* out = (float*)d_out;

  size_t off = 0;
  auto alloc = [&](size_t bytes) -> void* {
    void* p = (char*)d_ws + off;
    off += (bytes + 255) & ~(size_t)255;
    return p;
  };
  unsigned* keys  = (unsigned*)alloc((size_t)N_ROI * 4);
  int*      flag  = (int*)alloc((size_t)N_ROI * 4);
  unsigned* hist  = (unsigned*)alloc(4 * 256 * 4);
  int*      state = (int*)alloc(256);
  int*      idx   = (int*)alloc((size_t)TOPK * 4);
  float*    nq    = (float*)alloc((size_t)NHEAD * TOPK * 4);
  float*    nk    = (float*)alloc((size_t)NHEAD * TOPK * 4);
  _Float16* A2    = (_Float16*)alloc((size_t)TOPK * 2 * DIM * 2);  // [agg|dst]
  _Float16* Qh    = (_Float16*)alloc((size_t)TOPK * DIM * 2);
  _Float16* Kh    = (_Float16*)alloc((size_t)TOPK * DIM * 2);
  _Float16* Vh    = (_Float16*)alloc((size_t)TOPK * DIM * 2);
  float*    Obuf  = (float*)alloc((size_t)TOPK * DIM * 4);
  _Float16* wq16  = (_Float16*)alloc((size_t)DIM * DIM * 2);
  _Float16* wk16  = (_Float16*)alloc((size_t)DIM * DIM * 2);
  _Float16* wv16  = (_Float16*)alloc((size_t)DIM * DIM * 2);
  _Float16* wo16  = (_Float16*)alloc((size_t)DIM * 2 * DIM * 2);

  init_kernel<<<1, 256, 0, stream>>>(hist, state);
  scores_keys_kernel<<<N_ROI * 32 / 256, 256, 0, stream>>>(roi, w_sc, b_sc, keys, flag);
  for (int pass = 0; pass < 4; ++pass) {
    hist_kernel<<<N_ROI / 256, 256, 0, stream>>>(keys, hist, state, pass);
    scan_kernel<<<1, 32, 0, stream>>>(hist, state, pass);
  }
  compact_kernel<<<1, 1024, 0, stream>>>(keys, state, idx, flag);
  gather_kernel<<<TOPK * DIM / 4 / 256, 256, 0, stream>>>(roi, idx, A2);

  f32_to_f16_kernel<<<DIM * DIM / 4 / 256, 256, 0, stream>>>(wq, wq16, DIM * DIM / 4);
  f32_to_f16_kernel<<<DIM * DIM / 4 / 256, 256, 0, stream>>>(wk, wk16, DIM * DIM / 4);
  f32_to_f16_kernel<<<DIM * DIM / 4 / 256, 256, 0, stream>>>(wv, wv16, DIM * DIM / 4);
  f32_to_f16_kernel<<<DIM * 2 * DIM / 4 / 256, 256, 0, stream>>>(wo, wo16, DIM * 2 * DIM / 4);

  const _Float16* Xh = A2 + DIM;  // right half of A2, row stride 2*DIM
  dim3 gqkv(DIM / 128, TOPK / 128);
  gemm_nt_kernel<<<gqkv, 256, 0, stream>>>(Xh, 2 * DIM, wq16, bq, (void*)Qh, 0, DIM, DIM);
  gemm_nt_kernel<<<gqkv, 256, 0, stream>>>(Xh, 2 * DIM, wk16, bk, (void*)Kh, 0, DIM, DIM);
  gemm_nt_kernel<<<gqkv, 256, 0, stream>>>(Xh, 2 * DIM, wv16, bv, (void*)Vh, 0, DIM, DIM);

  norms_kernel<<<TOPK * NHEAD * 32 / 256, 256, 0, stream>>>(Qh, nq);
  norms_kernel<<<TOPK * NHEAD * 32 / 256, 256, 0, stream>>>(Kh, nk);

  flash_kernel<<<dim3(TOPK / 128, NHEAD), 256, 0, stream>>>(Qh, Kh, Vh, nq, nk, A2);

  gemm_nt_kernel<<<dim3(DIM / 128, TOPK / 128), 256, 0, stream>>>(
      A2, 2 * DIM, wo16, bo, (void*)Obuf, 1, DIM, 2 * DIM);

  finalize_kernel<<<N_ROI * DIM / 4 / 256, 256, 0, stream>>>(roi, flag, Obuf, out);
  (void)in_sizes; (void)n_in; (void)out_size; (void)ws_size;
}
